// MultiHeadAttention_18485539242887
// MI455X (gfx1250) — matrix operand order repaired and checked by reading, not compiled, not run on hardware
//
#include <hip/hip_runtime.h>
#include <hip/hip_bf16.h>

// ---------------------------------------------------------------------------
// MHA for MI455X (gfx1250, wave32, WMMA 16x16x32 f16 -> f32)
// ---------------------------------------------------------------------------

typedef _Float16 v16h __attribute__((ext_vector_type(16)));
typedef _Float16 v8h  __attribute__((ext_vector_type(8)));
typedef _Float16 v4h  __attribute__((ext_vector_type(4)));
typedef float    v8f  __attribute__((ext_vector_type(8)));

#define D_MODEL 1024
#define N_HEADS 16
#define HEAD_DIM 64
#define BATCH 2
#define SEQ 2048
#define MTOT (BATCH * SEQ)          // 4096
#define SCL  0.18033688f            // (1/sqrt(64)) * log2(e)

// ---- WMMA fragment loads per CDNA5 ISA VGPR layouts -----------------------
// A 16x32 (MxK), 16-bit: lane L -> row M = L&15.
//   halfs 0..7  = K (L<16 ? k0+0..7  : k0+8..15)
//   halfs 8..15 = K (L<16 ? k0+16..23: k0+24..31)
__device__ __forceinline__ v16h load_a_frag(const _Float16* __restrict__ base,
                                            int ldk, int lane, int k0) {
  const _Float16* p = base + (size_t)(lane & 15) * ldk + k0 + ((lane >> 4) << 3);
  v8h lo = *(const v8h*)p;
  v8h hi = *(const v8h*)(p + 16);
  v16h r;
#pragma unroll
  for (int i = 0; i < 8; ++i) { r[i] = lo[i]; r[i + 8] = hi[i]; }
  return r;
}

// B 32x16 (KxN), 16-bit: lane L -> column N = L&15.
//   halfs 0..7 = K (L<16 ? k0+0..7 : k0+8..15), halfs 8..15 = K (L<16 ? k0+16..23 : k0+24..31), contiguous in memory when
//   each N-column is a contiguous K-row (W row-major [NxK], K as [L x Dh],
//   or V stored transposed [Dh x L]).
__device__ __forceinline__ v16h load_b_frag(const _Float16* __restrict__ base,
                                            int ldk, int lane, int k0) {
  const _Float16* p = base + (size_t)(lane & 15) * ldk + k0 + ((lane >> 4) << 3);
  v8h lo = *(const v8h*)p;
  v8h hi = *(const v8h*)(p + 16);
  v16h r;
#pragma unroll
  for (int i = 0; i < 8; ++i) { r[i] = lo[i]; r[i + 8] = hi[i]; }
  return r;
}

__device__ __forceinline__ v8f wmma_f16(v16h a, v16h b, v8f c) {
  return __builtin_amdgcn_wmma_f32_16x16x32_f16(false, a, false, b,
                                                (short)0, c, false, false);
}

// ---- DPP16 butterfly reductions over 16-lane groups (pure VALU, no LDS) ---
// quad_perm(1,0,3,2)=0xB1 (xor1), quad_perm(2,3,0,1)=0x4E (xor2),
// row_half_mirror=0x141 (combines quads), row_mirror=0x140 (combines octets).
// DPP16 rows are 16 lanes, so the two halves of the wave reduce independently,
// matching the WMMA C/D fragment row mapping.
template <int CTRL>
__device__ __forceinline__ float dpp_movf(float x) {
  return __int_as_float(__builtin_amdgcn_update_dpp(
      0, __float_as_int(x), CTRL, 0xF, 0xF, true));
}
__device__ __forceinline__ float redmax16(float v) {
  v = fmaxf(v, dpp_movf<0x0B1>(v));
  v = fmaxf(v, dpp_movf<0x04E>(v));
  v = fmaxf(v, dpp_movf<0x141>(v));
  v = fmaxf(v, dpp_movf<0x140>(v));
  return v;
}
__device__ __forceinline__ float redsum16(float v) {
  v += dpp_movf<0x0B1>(v);
  v += dpp_movf<0x04E>(v);
  v += dpp_movf<0x141>(v);
  v += dpp_movf<0x140>(v);
  return v;
}

// ---------------------------------------------------------------------------
// f32 -> f16 conversion, 4-wide
// ---------------------------------------------------------------------------
__global__ __launch_bounds__(256) void cvt_f32_f16(const float* __restrict__ src,
                                                   _Float16* __restrict__ dst,
                                                   int n4) {
  int i = blockIdx.x * blockDim.x + threadIdx.x;
  if (i < n4) {
    float4 v = ((const float4*)src)[i];
    v4h h = {(_Float16)v.x, (_Float16)v.y, (_Float16)v.z, (_Float16)v.w};
    ((v4h*)dst)[i] = h;
  }
}

// ---------------------------------------------------------------------------
// GEMM: out = A[MxK] * W[NxK]^T + bias.  Wave computes 16x64 strip.
// MODE 0: f16 row-major    MODE 1: f16 -> [B,H,Dh,L] transpose   MODE 2: f32
// ---------------------------------------------------------------------------
template <int MODE>
__global__ __launch_bounds__(128) void gemm_wmma(const _Float16* __restrict__ A,
                                                 const _Float16* __restrict__ W,
                                                 const float* __restrict__ bias,
                                                 void* __restrict__ out,
                                                 int M, int N, int K) {
  const int lane = threadIdx.x & 31;
  const int wave = threadIdx.x >> 5;
  const int m0 = blockIdx.x * 16;
  const int n0 = (blockIdx.y * 4 + wave) * 64;
  const _Float16* Abase = A + (size_t)m0 * K;
  const _Float16* Arow = Abase + (size_t)(lane & 15) * K;

  v8f acc[4] = {{}, {}, {}, {}};
  for (int k0 = 0; k0 < K; k0 += 32) {
    if (k0 + 32 < K) {                 // prefetch next A strip (global_prefetch)
      __builtin_prefetch(Arow + k0 + 32, 0, 3);
    }
    v16h a = load_a_frag(Abase, K, lane, k0);
#pragma unroll
    for (int j = 0; j < 4; ++j) {
      v16h b = load_b_frag(W + (size_t)(n0 + j * 16) * K, K, lane, k0);
      acc[j] = wmma_f16(a, b, acc[j]);
    }
  }

  const int coll = lane & 15;
  const int rowoff = (lane >> 4) << 3;   // upper half lanes hold rows M+8..M+15
#pragma unroll
  for (int j = 0; j < 4; ++j) {
    const int Ncol = n0 + j * 16 + coll;
    const float bv = bias[Ncol];
#pragma unroll
    for (int r = 0; r < 8; ++r) {
      const int Mrow = m0 + r + rowoff;
      const float val = acc[j][r] + bv;
      if (MODE == 0) {
        ((_Float16*)out)[(size_t)Mrow * N + Ncol] = (_Float16)val;
      } else if (MODE == 1) {
        // (b,l,h,d) -> Vt[((b*16+h)*64+d)*2048 + l]
        const int bb = Mrow >> 11, l = Mrow & 2047;
        const int h = Ncol >> 6, d = Ncol & 63;
        ((_Float16*)out)[(((size_t)bb * N_HEADS + h) * HEAD_DIM + d) * SEQ + l] =
            (_Float16)val;
      } else {
        ((float*)out)[(size_t)Mrow * N + Ncol] = val;
      }
    }
  }
}

// ---------------------------------------------------------------------------
// Flash attention: one wave per 16-query tile per (b,h).
// Qh,Kh: f16 row-major [4096,1024] (head h at col h*64). Vt: f16 [B,H,Dh,L].
// Ctx out: f16 row-major [4096,1024].
// ---------------------------------------------------------------------------
__global__ __launch_bounds__(128) void attn_kernel(const _Float16* __restrict__ Qh,
                                                   const _Float16* __restrict__ Kh,
                                                   const _Float16* __restrict__ Vt,
                                                   _Float16* __restrict__ Ctx) {
  __shared__ _Float16 Pbuf[4][16 * 32];
  const int lane = threadIdx.x & 31;
  const int wave = threadIdx.x >> 5;
  const int bh = blockIdx.y;
  const int b = bh >> 4, h = bh & 15;
  const int q0 = (blockIdx.x * 4 + wave) * 16;

  const _Float16* Qbase = Qh + ((size_t)(b * SEQ + q0)) * D_MODEL + h * HEAD_DIM;
  const v16h aq0 = load_a_frag(Qbase, D_MODEL, lane, 0);
  const v16h aq1 = load_a_frag(Qbase, D_MODEL, lane, 32);

  const _Float16* Kbh = Kh + (size_t)(b * SEQ) * D_MODEL + h * HEAD_DIM;
  const _Float16* Vbh = Vt + (size_t)bh * HEAD_DIM * SEQ;
  _Float16* Pw = &Pbuf[wave][0];

  v8f o[4] = {{}, {}, {}, {}};
  float m_[8], l_[8];
#pragma unroll
  for (int r = 0; r < 8; ++r) { m_[r] = -3.0e38f; l_[r] = 0.0f; }

  for (int kv0 = 0; kv0 < SEQ; kv0 += 32) {
    // ---- scores S = Q * K^T for 32 keys (two 16-wide tiles) ----
    const _Float16* K0 = Kbh + (size_t)kv0 * D_MODEL;
    const _Float16* K1 = K0 + (size_t)16 * D_MODEL;
    v8f s0 = {}, s1 = {};
    s0 = wmma_f16(aq0, load_b_frag(K0, D_MODEL, lane, 0), s0);
    s0 = wmma_f16(aq1, load_b_frag(K0, D_MODEL, lane, 32), s0);
    s1 = wmma_f16(aq0, load_b_frag(K1, D_MODEL, lane, 0), s1);
    s1 = wmma_f16(aq1, load_b_frag(K1, D_MODEL, lane, 32), s1);

    // ---- online softmax (DPP16 reductions; rows live in 16-lane halves) ---
    float cf[8];
#pragma unroll
    for (int r = 0; r < 8; ++r) {
      float t = redmax16(fmaxf(s0[r], s1[r]));
      const float mn = fmaxf(m_[r], t);
      const float corr = exp2f((m_[r] - mn) * SCL);
      const float p0 = exp2f((s0[r] - mn) * SCL);
      const float p1 = exp2f((s1[r] - mn) * SCL);
      l_[r] = l_[r] * corr + redsum16(p0 + p1);
      m_[r] = mn;
      cf[r] = corr;
      const int row = r + ((lane >> 4) << 3);   // C/D layout -> P row index
      Pw[row * 32 + (lane & 15)] = (_Float16)p0;
      Pw[row * 32 + 16 + (lane & 15)] = (_Float16)p1;
    }
    asm volatile("s_wait_dscnt 0" ::: "memory");

    // ---- regather P in A-fragment layout from LDS ----
    const _Float16* pp = Pw + (size_t)(lane & 15) * 32 + ((lane >> 4) << 3);
    v8h lo = *(const v8h*)pp;
    v8h hi = *(const v8h*)(pp + 16);
    v16h ap;
#pragma unroll
    for (int i = 0; i < 8; ++i) { ap[i] = lo[i]; ap[i + 8] = hi[i]; }

    // ---- rescale O, accumulate P @ V ----
#pragma unroll
    for (int j = 0; j < 4; ++j) {
#pragma unroll
      for (int r = 0; r < 8; ++r) o[j][r] *= cf[r];
      v16h bv = load_b_frag(Vbh + (size_t)(j * 16) * SEQ, SEQ, lane, kv0);
      o[j] = wmma_f16(ap, bv, o[j]);
    }
  }

  // ---- normalize and store context ----
  const int coll = lane & 15;
  const int rowoff = (lane >> 4) << 3;
#pragma unroll
  for (int j = 0; j < 4; ++j) {
#pragma unroll
    for (int r = 0; r < 8; ++r) {
      const float val = o[j][r] / l_[r];
      const int row = q0 + r + rowoff;
      const int col = h * HEAD_DIM + j * 16 + coll;
      Ctx[(size_t)(b * SEQ + row) * D_MODEL + col] = (_Float16)val;
    }
  }
}

// ---------------------------------------------------------------------------
// Host-side launch
// ---------------------------------------------------------------------------
extern "C" void kernel_launch(void* const* d_in, const int* in_sizes, int n_in,
                              void* d_out, int out_size, void* d_ws, size_t ws_size,
                              hipStream_t stream) {
  const float* q  = (const float*)d_in[0];
  const float* k  = (const float*)d_in[1];
  const float* v  = (const float*)d_in[2];
  const float* Wq = (const float*)d_in[3];
  const float* bq = (const float*)d_in[4];
  const float* Wk = (const float*)d_in[5];
  const float* bk = (const float*)d_in[6];
  const float* Wv = (const float*)d_in[7];
  const float* bv = (const float*)d_in[8];
  const float* Wo = (const float*)d_in[9];
  const float* bo = (const float*)d_in[10];

  char* ws = (char*)d_ws;
  const size_t SZ_X = (size_t)MTOT * D_MODEL * sizeof(_Float16);    // 8 MB
  const size_t SZ_W = (size_t)D_MODEL * D_MODEL * sizeof(_Float16); // 2 MB
  _Float16* qh  = (_Float16*)(ws);                       // converted inputs
  _Float16* kh  = (_Float16*)(ws + SZ_X);
  _Float16* vh  = (_Float16*)(ws + 2 * SZ_X);
  _Float16* WqH = (_Float16*)(ws + 3 * SZ_X);
  _Float16* WkH = (_Float16*)(ws + 3 * SZ_X + SZ_W);
  _Float16* WvH = (_Float16*)(ws + 3 * SZ_X + 2 * SZ_W);
  _Float16* WoH = (_Float16*)(ws + 3 * SZ_X + 3 * SZ_W);
  _Float16* Qrm = (_Float16*)(ws + 3 * SZ_X + 4 * SZ_W);           // Q proj
  _Float16* Krm = (_Float16*)(ws + 4 * SZ_X + 4 * SZ_W);           // K proj
  _Float16* Vt  = (_Float16*)(ws + 5 * SZ_X + 4 * SZ_W);           // V^T proj
  _Float16* Ctx = (_Float16*)(ws + 6 * SZ_X + 4 * SZ_W);           // attn out

  const int nX4 = (MTOT * D_MODEL) / 4;       // 1048576
  const int nW4 = (D_MODEL * D_MODEL) / 4;    // 262144
  cvt_f32_f16<<<nX4 / 256, 256, 0, stream>>>(q, qh, nX4);
  cvt_f32_f16<<<nX4 / 256, 256, 0, stream>>>(k, kh, nX4);
  cvt_f32_f16<<<nX4 / 256, 256, 0, stream>>>(v, vh, nX4);
  cvt_f32_f16<<<nW4 / 256, 256, 0, stream>>>(Wq, WqH, nW4);
  cvt_f32_f16<<<nW4 / 256, 256, 0, stream>>>(Wk, WkH, nW4);
  cvt_f32_f16<<<nW4 / 256, 256, 0, stream>>>(Wv, WvH, nW4);
  cvt_f32_f16<<<nW4 / 256, 256, 0, stream>>>(Wo, WoH, nW4);

  dim3 ggrid(MTOT / 16, D_MODEL / 256);       // 256 x 4, 128 threads (4 waves)
  gemm_wmma<0><<<ggrid, 128, 0, stream>>>(qh, WqH, bq, Qrm, MTOT, D_MODEL, D_MODEL);
  gemm_wmma<0><<<ggrid, 128, 0, stream>>>(kh, WkH, bk, Krm, MTOT, D_MODEL, D_MODEL);
  gemm_wmma<1><<<ggrid, 128, 0, stream>>>(vh, WvH, bv, Vt,  MTOT, D_MODEL, D_MODEL);

  dim3 agrid(SEQ / 16 / 4, BATCH * N_HEADS);  // 32 x 32, 128 threads (4 waves)
  attn_kernel<<<agrid, 128, 0, stream>>>(Qrm, Krm, Vt, Ctx);

  gemm_wmma<2><<<ggrid, 128, 0, stream>>>(Ctx, WoH, bo, d_out, MTOT, D_MODEL, D_MODEL);
}
